// VRNN_19439021982217
// MI455X (gfx1250) — matrix operand order repaired and checked by reading, not compiled, not run on hardware
//
#include <hip/hip_runtime.h>
#include <hip/hip_bf16.h>
#include <math.h>

// MI455X / gfx1250 tree-VRNN.
//  - WMMA (v_wmma_f32_16x16x32_f16) for all large GEMMs.
//  - Vocab/input-gate GEMMs: B-slice double-buffered in LDS via the CDNA5
//    async path (global_load_async_to_lds_b128 + s_wait_asynccnt), hiding
//    global latency behind WMMA issues and killing 8x redundant L2 reads.
//  - Recurrence: GRU rows are independent -> 8 workgroups, one 16-row d-tile
//    each, no inter-block sync; h tile resident in LDS; 12 WMMA accumulators
//    per wave; gate math in-register.

typedef __attribute__((ext_vector_type(16))) _Float16 v16h;
typedef __attribute__((ext_vector_type(8)))  _Float16 v8h;
typedef __attribute__((ext_vector_type(8)))  float    v8f;

#define T_DIM 32
#define D_DIM 128
#define H_DIM 256
#define V_DIM 32000
#define Z_DIM 256
#define G_DIM (3 * H_DIM)   // 768 gate columns (r,z,n)

// ---------------------------------------------------------------- helpers ---

__device__ __forceinline__ v8f vzero8f() {
  v8f z;
#pragma unroll
  for (int i = 0; i < 8; ++i) z[i] = 0.0f;
  return z;
}

__device__ __forceinline__ v8f wmma16(v16h a, v16h b, v8f c) {
  // D = A(16x32 f16) * B(32x16 f16) + C(16x16 f32)
  return __builtin_amdgcn_wmma_f32_16x16x32_f16(
      /*neg_a=*/false, a, /*neg_b=*/false, b,
      /*c_mod=*/(short)0, c, /*reuse_a=*/false, /*reuse_b=*/false);
}

// CDNA5 async copy: 16B global -> LDS, tracked with ASYNCcnt (no VGPR round
// trip). VDST = 32-bit LDS byte offset (low 32 bits of a generic pointer to
// __shared__ are the wave-relative LDS address), VADDR = 64-bit global addr.
__device__ __forceinline__ void async_g2l_b128(const void* gaddr, void* laddr) {
  asm volatile("global_load_async_to_lds_b128 %0, %1, off"
               :
               : "v"((unsigned)(size_t)laddr), "v"(gaddr)
               : "memory");
}
__device__ __forceinline__ void wait_asynccnt0() {
  asm volatile("s_wait_asynccnt 0x0" ::: "memory");
}

// A operand from row-major f16 [M x lda] at tile (m0,k0).
// ISA 16-bit A layout: lane<16 -> row m0+lane, halfs {K0..7, K16..23};
//                      lane>=16 -> row m0+lane-16, halfs {K8..15, K24..31}.
__device__ __forceinline__ v16h load_a_g(const _Float16* __restrict__ A,
                                         int lda, int m0, int k0, int lane) {
  const _Float16* p =
      A + (size_t)(m0 + (lane & 15)) * lda + (k0 + ((lane >> 4) << 3));
  union { v16h v; v8h h[2]; } u;
  u.h[0] = *(const v8h*)(p);        // K offset +0 (or +8)
  u.h[1] = *(const v8h*)(p + 16);   // K offset +16 (or +24)
  return u.v;
}

// B operand of A*W^T: W is row-major f16 [N x ldb] (K contiguous), tile (n0,k0).
// ISA 16-bit B layout: lane<16 -> col n0+lane, halfs {K0..7, K16..23};
//                      lane>=16 -> col n0+lane-16, halfs {K8..15, K24..31}.
__device__ __forceinline__ v16h load_b_g(const _Float16* __restrict__ W,
                                         int ldb, int n0, int k0, int lane) {
  const _Float16* p =
      W + (size_t)(n0 + (lane & 15)) * ldb + (k0 + ((lane >> 4) << 3));
  union { v16h v; v8h h[2]; } u;
  u.h[0] = *(const v8h*)(p);
  u.h[1] = *(const v8h*)(p + 16);
  return u.v;
}

// ---------------------------------------------------------------- kernels ---

__global__ void f32_to_f16_kernel(const float* __restrict__ src,
                                  _Float16* __restrict__ dst, int n) {
  int i = blockIdx.x * blockDim.x + threadIdx.x;
  if (i < n) dst[i] = (_Float16)src[i];
}

__global__ void gather_emb_kernel(const int* __restrict__ nodes,
                                  const float* __restrict__ emb,
                                  _Float16* __restrict__ X16, int rows, int h) {
  int i = blockIdx.x * blockDim.x + threadIdx.x;
  if (i >= rows * h) return;
  int r = i / h, c = i - r * h;
  X16[i] = (_Float16)emb[(size_t)nodes[r] * h + c];
}

// C[M,N] = A16[M,K] * W16[N,K]^T + bias[N], f32 out.
// Requires M%128==0, N%64==0, K%32==0 (all call sites satisfy this).
// Block = 256 threads (8 waves); block tile = 128 M x 64 N. The 64x32 f16
// B-slice is double-buffered in LDS: slice k0+32 is copied with
// global_load_async_to_lds_b128 while WMMA consumes slice k0, then
// s_wait_asynccnt 0 + barrier flips buffers.
__global__ __launch_bounds__(256) void gemm_bias_wmma(
    const _Float16* __restrict__ A, const _Float16* __restrict__ W,
    const float* __restrict__ bias, float* __restrict__ C,
    int M, int N, int K) {
  __shared__ _Float16 Bs[2][64][40];  // rows padded to 80B: 16B-aligned, bank-safe
  const int lane = threadIdx.x & 31;
  const int wave = threadIdx.x >> 5;
  const int m0 = (blockIdx.y * 8 + wave) << 4;
  const int n0 = blockIdx.x << 6;

  const int snl = threadIdx.x >> 2;        // staging: n_local 0..63
  const int skk = (threadIdx.x & 3) << 3;  // staging: kk 0,8,16,24
  const _Float16* wrow = W + (size_t)(n0 + snl) * K + skk;  // this thread's 16B lane

  v8f acc[4];
#pragma unroll
  for (int i = 0; i < 4; ++i) acc[i] = vzero8f();

  // prologue: async-stage k-slice 0 into buffer 0
  async_g2l_b128(wrow, &Bs[0][snl][skk]);
  wait_asynccnt0();
  __syncthreads();

  int buf = 0;
  for (int k0 = 0; k0 < K; k0 += 32) {
    if (k0 + 32 < K)   // issue next slice into the other buffer (no wait yet)
      async_g2l_b128(wrow + k0 + 32, &Bs[buf ^ 1][snl][skk]);

    v16h a = load_a_g(A, K, m0, k0, lane);
#pragma unroll
    for (int nt = 0; nt < 4; ++nt) {
      // B fragment from LDS: col = nt*16 + (lane&15), K = (lane>=16)*8 + {0..7, 16..23}
      const _Float16* p = &Bs[buf][nt * 16 + (lane & 15)][(lane >> 4) << 3];
      union { v16h v; v8h h[2]; } u;
      u.h[0] = *(const v8h*)(p);
      u.h[1] = *(const v8h*)(p + 16);
      acc[nt] = wmma16(a, u.v, acc[nt]);
    }
    wait_asynccnt0();   // next-slice copies landed
    __syncthreads();    // and everyone is done reading Bs[buf]
    buf ^= 1;
  }
  // C/D layout: reg r, lane -> M = r + 8*(lane>=16), N = lane&15
  const int nc = lane & 15;
  const int mb = m0 + ((lane >> 4) << 3);
#pragma unroll
  for (int nt = 0; nt < 4; ++nt) {
    const int n = n0 + nt * 16 + nc;
    const float bv = bias[n];
#pragma unroll
    for (int r = 0; r < 8; ++r)
      C[(size_t)(mb + r) * N + n] = acc[nt][r] + bv;
  }
}

// Branch-aware GRU scan. GRU rows are independent across the batch dim D, so
// grid = 8 blocks, each owning a 16-row d-tile through all T steps with NO
// inter-block synchronization. Block = 256 threads (8 waves); each wave owns
// 2 j-tiles and keeps 12 gate accumulators (2 j x 2 cells x r/z/n) live so the
// A fragment (h tile, LDS f32 -> f16) is built once per k-slice and reused
// across 12 v_wmma issues. Per step each WGP reads the Whh pair exactly once
// (each fragment consumed by exactly one wave -> direct global loads, no LDS
// staging hop).
__global__ __launch_bounds__(256, 1) void tree_gru_kernel(
    const float* __restrict__ Gi_c, const float* __restrict__ Gi_s,   // [T*D,768]
    const _Float16* __restrict__ Whh_c, const _Float16* __restrict__ Whh_s, // [768,256] f16
    const float* __restrict__ bhh_c, const float* __restrict__ bhh_s, // [768]
    const unsigned char* __restrict__ edges,                          // [T*D] bool
    const float* __restrict__ h_init,                                 // [128,256]
    _Float16* __restrict__ outs16,                                    // [T*128,256] or null
    float* __restrict__ h_final)                                      // [128,256]
{
  __shared__ float hsh[16][H_DIM + 1];   // this block's 16 h rows, row-padded
  const int tid = threadIdx.x;
  const int lane = tid & 31;
  const int wave = tid >> 5;
  const int lhi = lane >> 4;
  const int ln15 = lane & 15;
  const int dbase = blockIdx.x << 4;     // first global row of this d-tile

  for (int i = tid; i < 16 * H_DIM; i += 256)
    hsh[i >> 8][i & 255] = h_init[(size_t)(dbase + (i >> 8)) * H_DIM + (i & 255)];
  __syncthreads();

  for (int t = 0; t < T_DIM; ++t) {
    v8f acc[12];
#pragma unroll
    for (int g = 0; g < 12; ++g) acc[g] = vzero8f();

    for (int k0 = 0; k0 < H_DIM; k0 += 32) {
      // A fragment: local h rows 0..15 (f32 LDS -> f16, per ISA A layout)
      union { v16h v; _Float16 e[16]; } a;
      {
        const int kb = k0 + (lhi << 3);
#pragma unroll
        for (int i = 0; i < 8; ++i) {
          a.e[i]     = (_Float16)hsh[ln15][kb + i];
          a.e[i + 8] = (_Float16)hsh[ln15][kb + 16 + i];
        }
      }
#pragma unroll
      for (int jj = 0; jj < 2; ++jj) {
        const int j0 = ((wave << 1) + jj) << 4;
#pragma unroll
        for (int g = 0; g < 3; ++g) {
          v16h bc = load_b_g(Whh_c, H_DIM, g * H_DIM + j0, k0, lane);
          acc[jj * 6 + g] = wmma16(a.v, bc, acc[jj * 6 + g]);
          v16h bs = load_b_g(Whh_s, H_DIM, g * H_DIM + j0, k0, lane);
          acc[jj * 6 + 3 + g] = wmma16(a.v, bs, acc[jj * 6 + 3 + g]);
        }
      }
    }
    // In-register GRU gate math.
    float hn[16];
#pragma unroll
    for (int jj = 0; jj < 2; ++jj) {
      const int j = (((wave << 1) + jj) << 4) + ln15;
      const float bcr = bhh_c[j], bcz = bhh_c[H_DIM + j], bcn = bhh_c[2 * H_DIM + j];
      const float bsr = bhh_s[j], bsz = bhh_s[H_DIM + j], bsn = bhh_s[2 * H_DIM + j];
      const int mb = lhi << 3;
#pragma unroll
      for (int r = 0; r < 8; ++r) {
        const int dloc = mb + r;
        const int d = dbase + dloc;
        const size_t row = (size_t)(t * D_DIM + d) * G_DIM;
        const float hprev = hsh[dloc][j];
        const float rc = 1.0f / (1.0f + expf(-(Gi_c[row + j] + acc[jj * 6 + 0][r] + bcr)));
        const float zc = 1.0f / (1.0f + expf(-(Gi_c[row + H_DIM + j] + acc[jj * 6 + 1][r] + bcz)));
        const float nc = tanhf(Gi_c[row + 2 * H_DIM + j] + rc * (acc[jj * 6 + 2][r] + bcn));
        const float hc = (1.0f - zc) * nc + zc * hprev;
        const float rs = 1.0f / (1.0f + expf(-(Gi_s[row + j] + acc[jj * 6 + 3][r] + bsr)));
        const float zs = 1.0f / (1.0f + expf(-(Gi_s[row + H_DIM + j] + acc[jj * 6 + 4][r] + bsz)));
        const float ns = tanhf(Gi_s[row + 2 * H_DIM + j] + rs * (acc[jj * 6 + 5][r] + bsn));
        const float hsv = (1.0f - zs) * ns + zs * hprev;
        hn[jj * 8 + r] = edges[t * D_DIM + d] ? hc : hsv;
      }
    }
    __syncthreads();   // all reads of h done
#pragma unroll
    for (int jj = 0; jj < 2; ++jj) {
      const int j = (((wave << 1) + jj) << 4) + ln15;
      const int mb = lhi << 3;
#pragma unroll
      for (int r = 0; r < 8; ++r) {
        const int dloc = mb + r;
        const float v = hn[jj * 8 + r];
        hsh[dloc][j] = v;
        if (outs16)
          outs16[(size_t)(t * D_DIM + dbase + dloc) * H_DIM + j] = (_Float16)v;
      }
    }
    __syncthreads();   // h updated for next step
  }
  for (int i = tid; i < 16 * H_DIM; i += 256)
    h_final[(size_t)(dbase + (i >> 8)) * H_DIM + (i & 255)] = hsh[i >> 8][i & 255];
}

// Small exact f32 linear: Y[M,N] = X[M,K] * W[N,K]^T + b[N]
__global__ void linear_f32_kernel(const float* __restrict__ X,
                                  const float* __restrict__ W,
                                  const float* __restrict__ b,
                                  float* __restrict__ Y, int M, int N, int K) {
  int idx = blockIdx.x * blockDim.x + threadIdx.x;
  if (idx >= M * N) return;
  int m = idx / N, n = idx - m * N;
  float s = b[n];
  const float* x = X + (size_t)m * K;
  const float* w = W + (size_t)n * K;
  for (int k = 0; k < K; ++k) s += x[k] * w[k];
  Y[idx] = s;
}

__global__ void reparam_kernel(const float* __restrict__ eps,
                               const float* __restrict__ mean,
                               const float* __restrict__ logv,
                               float* __restrict__ z, int n) {
  int i = blockIdx.x * blockDim.x + threadIdx.x;
  if (i < n) z[i] = eps[i] * expf(0.5f * logv[i]) + mean[i];
}

// In-place log_softmax over each length-N row; one block per row.
__global__ __launch_bounds__(256) void log_softmax_rows(float* __restrict__ logits, int N) {
  __shared__ float red[256];
  float* p = logits + (size_t)blockIdx.x * N;
  const int tx = threadIdx.x;
  float mx = -1e30f;
  for (int i = tx; i < N; i += 256) mx = fmaxf(mx, p[i]);
  red[tx] = mx; __syncthreads();
  for (int s = 128; s > 0; s >>= 1) {
    if (tx < s) red[tx] = fmaxf(red[tx], red[tx + s]);
    __syncthreads();
  }
  mx = red[0]; __syncthreads();
  float sum = 0.0f;
  for (int i = tx; i < N; i += 256) sum += expf(p[i] - mx);
  red[tx] = sum; __syncthreads();
  for (int s = 128; s > 0; s >>= 1) {
    if (tx < s) red[tx] += red[tx + s];
    __syncthreads();
  }
  const float lse = mx + logf(red[0]);
  for (int i = tx; i < N; i += 256) p[i] = p[i] - lse;
}

// ----------------------------------------------------------------- launch ---

extern "C" void kernel_launch(void* const* d_in, const int* in_sizes, int n_in,
                              void* d_out, int out_size, void* d_ws, size_t ws_size,
                              hipStream_t stream) {
  (void)in_sizes; (void)n_in; (void)out_size; (void)ws_size;
  const int*           nodes = (const int*)d_in[0];
  const unsigned char* edges = (const unsigned char*)d_in[1];   // jnp bool_
  const float*         emb   = (const float*)d_in[2];
  const float* Wih[4] = {(const float*)d_in[3], (const float*)d_in[7],
                         (const float*)d_in[11], (const float*)d_in[15]};
  const float* Whh[4] = {(const float*)d_in[4], (const float*)d_in[8],
                         (const float*)d_in[12], (const float*)d_in[16]};
  const float* bih[4] = {(const float*)d_in[5], (const float*)d_in[9],
                         (const float*)d_in[13], (const float*)d_in[17]};
  const float* bhh[4] = {(const float*)d_in[6], (const float*)d_in[10],
                         (const float*)d_in[14], (const float*)d_in[18]};
  const float* W_mean = (const float*)d_in[19]; const float* b_mean = (const float*)d_in[20];
  const float* W_logv = (const float*)d_in[21]; const float* b_logv = (const float*)d_in[22];
  const float* W_l2h  = (const float*)d_in[23]; const float* b_l2h  = (const float*)d_in[24];
  const float* W_out  = (const float*)d_in[25]; const float* b_out  = (const float*)d_in[26];
  const float* enc_init = (const float*)d_in[27];
  const float* eps      = (const float*)d_in[28];
  float* out = (float*)d_out;

  // workspace carve-up (~72 MB)
  char* wsp = (char*)d_ws;
  size_t off = 0;
  auto take = [&](size_t bytes) -> char* {
    char* p = wsp + off;
    off += (bytes + 255) & ~(size_t)255;
    return p;
  };
  _Float16 *Wih16[4], *Whh16[4];
  for (int i = 0; i < 4; ++i) {
    Wih16[i] = (_Float16*)take((size_t)G_DIM * H_DIM * sizeof(_Float16));
    Whh16[i] = (_Float16*)take((size_t)G_DIM * H_DIM * sizeof(_Float16));
  }
  _Float16* Wout16 = (_Float16*)take((size_t)V_DIM * H_DIM * sizeof(_Float16));
  _Float16* X16    = (_Float16*)take((size_t)T_DIM * D_DIM * H_DIM * sizeof(_Float16));
  float* Gi[4];
  for (int i = 0; i < 4; ++i)
    Gi[i] = (float*)take((size_t)T_DIM * D_DIM * G_DIM * sizeof(float));
  float* hid    = (float*)take((size_t)D_DIM * H_DIM * sizeof(float));
  float* h0     = (float*)take((size_t)D_DIM * H_DIM * sizeof(float));
  float* hdump  = (float*)take((size_t)D_DIM * H_DIM * sizeof(float));
  _Float16* outs16 = (_Float16*)take((size_t)T_DIM * D_DIM * H_DIM * sizeof(_Float16));

  const int TD = T_DIM * D_DIM;   // 4096

  // 1) f32 -> f16 weight conversions (layout preserved: [N,K] rows = B operand)
  for (int i = 0; i < 4; ++i) {
    f32_to_f16_kernel<<<(G_DIM * H_DIM + 255) / 256, 256, 0, stream>>>(Wih[i], Wih16[i], G_DIM * H_DIM);
    f32_to_f16_kernel<<<(G_DIM * H_DIM + 255) / 256, 256, 0, stream>>>(Whh[i], Whh16[i], G_DIM * H_DIM);
  }
  f32_to_f16_kernel<<<(int)(((size_t)V_DIM * H_DIM + 255) / 256), 256, 0, stream>>>(W_out, Wout16, V_DIM * H_DIM);

  // 2) token gather -> f16 activations X[T*D, H]
  gather_emb_kernel<<<(TD * H_DIM + 255) / 256, 256, 0, stream>>>(nodes, emb, X16, TD, H_DIM);

  // 3) hoisted input-gate GEMMs: Gi = X @ W_ih^T + b_ih  (4 cells, WMMA)
  for (int i = 0; i < 4; ++i)
    gemm_bias_wmma<<<dim3(G_DIM / 64, TD / 128), 256, 0, stream>>>(
        X16, Wih16[i], bih[i], Gi[i], TD, G_DIM, H_DIM);

  // 4) encoder scan: 8 independent d-tile blocks, WMMA recurrence
  tree_gru_kernel<<<D_DIM / 16, 256, 0, stream>>>(
      Gi[0], Gi[1], Whh16[0], Whh16[1], bhh[0], bhh[1], edges, enc_init,
      (_Float16*)nullptr, hid);

  // 5) VAE head (exact f32): mean/logv/z land in d_out tail in return order
  const size_t LOGP = (size_t)TD * V_DIM;
  float* meanp = out + LOGP;
  float* logvp = meanp + (size_t)D_DIM * Z_DIM;
  float* zp    = logvp + (size_t)D_DIM * Z_DIM;
  linear_f32_kernel<<<(D_DIM * Z_DIM + 255) / 256, 256, 0, stream>>>(hid, W_mean, b_mean, meanp, D_DIM, Z_DIM, H_DIM);
  linear_f32_kernel<<<(D_DIM * Z_DIM + 255) / 256, 256, 0, stream>>>(hid, W_logv, b_logv, logvp, D_DIM, Z_DIM, H_DIM);
  reparam_kernel<<<(D_DIM * Z_DIM + 255) / 256, 256, 0, stream>>>(eps, meanp, logvp, zp, D_DIM * Z_DIM);
  linear_f32_kernel<<<(D_DIM * H_DIM + 255) / 256, 256, 0, stream>>>(zp, W_l2h, b_l2h, h0, D_DIM, H_DIM, Z_DIM);

  // 6) decoder scan (emits per-step states as f16 for the vocab GEMM)
  tree_gru_kernel<<<D_DIM / 16, 256, 0, stream>>>(
      Gi[2], Gi[3], Whh16[2], Whh16[3], bhh[2], bhh[3], edges, h0,
      outs16, hdump);

  // 7) vocab projection (dominant GEMM, WMMA + async double-buffered B)
  gemm_bias_wmma<<<dim3(V_DIM / 64, TD / 128), 256, 0, stream>>>(
      outs16, Wout16, b_out, out, TD, V_DIM, H_DIM);
  log_softmax_rows<<<TD, 256, 0, stream>>>(out, V_DIM);
}
